// Model_32710470926482
// MI455X (gfx1250) — compile-verified, matrix operand order checked by reading, not run on hardware
//
#include <hip/hip_runtime.h>
#include <hip/hip_bf16.h>

typedef __attribute__((ext_vector_type(16))) _Float16 v16h;
typedef __attribute__((ext_vector_type(8)))  _Float16 v8h;
typedef __attribute__((ext_vector_type(8)))  float    v8f;

#define MROWS   131072          // B*N
#define NPTS    32768           // points per batch
#define HDIM    256
#define KDIM    256

__device__ __forceinline__ float gelu_f(float x) {
    float x3 = x * x * x;
    return 0.5f * x * (1.0f + tanhf(0.7978845608028654f * (x + 0.044715f * x3)));
}

// ---------------------------------------------------------------------------
// Preprocess: h = gelu(concat(x,fx) @ W1 + b1) @ W2 + b2 + placeholder
// ---------------------------------------------------------------------------
__global__ void k_pre(const float* __restrict__ X, const float* __restrict__ FX,
                      const float* __restrict__ W1, const float* __restrict__ B1,
                      const float* __restrict__ W2, const float* __restrict__ B2,
                      const float* __restrict__ ph, float* __restrict__ h) {
    __shared__ float u[512];
    int m = blockIdx.x;
    float x0 = X[(size_t)m * 2], x1 = X[(size_t)m * 2 + 1], f0 = FX[m];
    int t = threadIdx.x;
    for (int j = t; j < 512; j += 256) {
        float v = x0 * W1[j] + x1 * W1[512 + j] + f0 * W1[1024 + j] + B1[j];
        u[j] = gelu_f(v);
    }
    __syncthreads();
    float acc = B2[t];
    for (int k = 0; k < 512; ++k) acc += u[k] * W2[k * 256 + t];
    h[(size_t)m * 256 + t] = acc + ph[t];
}

// ---------------------------------------------------------------------------
// Weight convert + transpose: WT[n*K+k] = (f16) W[k*N+n]   (256x256)
// ---------------------------------------------------------------------------
__global__ void k_wt(const float* __restrict__ W, _Float16* __restrict__ WT) {
    int idx = blockIdx.x * 256 + threadIdx.x;   // 65536
    int n = idx >> 8, k = idx & 255;
    WT[idx] = (_Float16)W[k * 256 + n];
}

// ---------------------------------------------------------------------------
// LayerNorm over H=256 -> f16 output (feeds WMMA GEMMs).
// ---------------------------------------------------------------------------
__global__ void k_ln(const float* __restrict__ x, const float* __restrict__ g,
                     const float* __restrict__ b, _Float16* __restrict__ y) {
    int wave = threadIdx.x >> 5, lane = threadIdx.x & 31;
    int m = blockIdx.x * 8 + wave;
    const float* row = x + (size_t)m * 256;
    float v[8];
    float s = 0.f;
#pragma unroll
    for (int i = 0; i < 8; ++i) { v[i] = row[i * 32 + lane]; s += v[i]; }
#pragma unroll
    for (int off = 16; off > 0; off >>= 1) s += __shfl_xor(s, off, 32);
    float mu = s * (1.f / 256.f);
    float q = 0.f;
#pragma unroll
    for (int i = 0; i < 8; ++i) { float d = v[i] - mu; q += d * d; }
#pragma unroll
    for (int off = 16; off > 0; off >>= 1) q += __shfl_xor(q, off, 32);
    float rs = rsqrtf(q * (1.f / 256.f) + 1e-5f);
    _Float16* yo = y + (size_t)m * 256;
#pragma unroll
    for (int i = 0; i < 8; ++i) {
        int col = i * 32 + lane;
        yo[col] = (_Float16)((v[i] - mu) * rs * g[col] + b[col]);
    }
}

// ---------------------------------------------------------------------------
// WMMA GEMM with async-to-LDS double buffering.
// C[M x 256] = act(A16[M x 256] @ WT^T + bias) (+ residual)
// Block = 256 thr (8 waves); block tile 128x64; wave = 16 rows x 64 cols
// (4 WMMA accumulators). Tiles staged with GLOBAL_LOAD_ASYNC_TO_LDS_B128,
// pipelined with s_wait_asynccnt 3 (next stage's 3 chunks stay in flight).
// Fragment layout per CDNA5 16-bit 16x32 spec: halves [hi*8..+7] -> vgpr 0..3,
// halves [16+hi*8..+7] -> vgpr 4..7, so each fragment = 2x ds_load_b128.
// ---------------------------------------------------------------------------
__global__ void k_gemm(const _Float16* __restrict__ A16, const _Float16* __restrict__ WT,
                       const float* __restrict__ bias, const float* resid,
                       float* Cf, _Float16* Ch, int act) {
    __shared__ _Float16 At[2][128 * 32];
    __shared__ _Float16 Bt[2][64 * 32];
    int t    = threadIdx.x;
    int wave = t >> 5, lane = t & 31;
    int hi   = lane >> 4, l16 = lane & 15;
    int rowBase = blockIdx.x * 128;
    int colBase = blockIdx.y * 64;

    v8f acc[4];
#pragma unroll
    for (int i = 0; i < 4; ++i) acc[i] = (v8f){0, 0, 0, 0, 0, 0, 0, 0};

    const int ia0 = t >> 2,          ja0 = (t & 3) * 8;          // A chunk 0
    const int ia1 = (t + 256) >> 2,  ja1 = ((t + 256) & 3) * 8;  // A chunk 1
    const int nb  = t >> 2,          jb  = (t & 3) * 8;          // B chunk

    const _Float16* gA0 = &A16[(size_t)(rowBase + ia0) * KDIM + ja0];
    const _Float16* gA1 = &A16[(size_t)(rowBase + ia1) * KDIM + ja1];
    const _Float16* gB  = &WT[(size_t)(colBase + nb) * KDIM + jb];

    unsigned la0[2], la1[2], lbb[2];
#pragma unroll
    for (int bfr = 0; bfr < 2; ++bfr) {
        la0[bfr] = (unsigned)(uintptr_t)&At[bfr][ia0 * 32 + ja0];
        la1[bfr] = (unsigned)(uintptr_t)&At[bfr][ia1 * 32 + ja1];
        lbb[bfr] = (unsigned)(uintptr_t)&Bt[bfr][nb * 32 + jb];
    }

    auto issue = [&](int bfr, int kk) {
        asm volatile("global_load_async_to_lds_b128 %0, %1, off"
                     :: "v"(la0[bfr]),
                        "v"((unsigned long long)(uintptr_t)(gA0 + kk)) : "memory");
        asm volatile("global_load_async_to_lds_b128 %0, %1, off"
                     :: "v"(la1[bfr]),
                        "v"((unsigned long long)(uintptr_t)(gA1 + kk)) : "memory");
        asm volatile("global_load_async_to_lds_b128 %0, %1, off"
                     :: "v"(lbb[bfr]),
                        "v"((unsigned long long)(uintptr_t)(gB + kk)) : "memory");
    };

    issue(0, 0);
    for (int kk = 0; kk < KDIM; kk += 32) {
        int cur = (kk >> 5) & 1;
        issue(cur ^ 1, (kk + 32) & 255);     // prefetch next stage (tail wraps harmlessly)
        asm volatile("s_wait_asynccnt 0x3" ::: "memory");  // current stage landed
        __syncthreads();

        const _Float16* ap = &At[cur][(wave * 16 + l16) * 32 + hi * 8];
        v8h alo = *(const v8h*)ap;
        v8h ahi = *(const v8h*)(ap + 16);
        v16h af = __builtin_shufflevector(alo, ahi, 0, 1, 2, 3, 4, 5, 6, 7,
                                          8, 9, 10, 11, 12, 13, 14, 15);
#pragma unroll
        for (int tt = 0; tt < 4; ++tt) {
            const _Float16* bp = &Bt[cur][(tt * 16 + l16) * 32 + hi * 8];
            v8h blo = *(const v8h*)bp;
            v8h bhi = *(const v8h*)(bp + 16);
            v16h bf = __builtin_shufflevector(blo, bhi, 0, 1, 2, 3, 4, 5, 6, 7,
                                              8, 9, 10, 11, 12, 13, 14, 15);
            acc[tt] = __builtin_amdgcn_wmma_f32_16x16x32_f16(false, af, false, bf,
                                                             (short)0, acc[tt],
                                                             false, false);
        }
        __syncthreads();
    }

#pragma unroll
    for (int tt = 0; tt < 4; ++tt) {
        int n = colBase + tt * 16 + l16;
        float bn = bias[n];
#pragma unroll
        for (int v = 0; v < 8; ++v) {
            int mo = rowBase + wave * 16 + v + hi * 8;
            float x = acc[tt][v] + bn;
            if (resid) x += resid[(size_t)mo * 256 + n];
            if (act) x = gelu_f(x);
            if (Cf) Cf[(size_t)mo * 256 + n] = x;
            if (Ch) Ch[(size_t)mo * 256 + n] = (_Float16)x;
        }
    }
}

// ---------------------------------------------------------------------------
// Slice softmax + slice-token accumulation.
// Block = 8 waves = 8 heads; each block covers 1024 points of one batch.
// ---------------------------------------------------------------------------
__global__ void k_slice(const float* __restrict__ xm, const float* __restrict__ fxm,
                        const float* __restrict__ wsl, const float* __restrict__ bsl,
                        const float* __restrict__ temp,
                        float* __restrict__ sw, float* __restrict__ norm,
                        float* __restrict__ st) {
    int hh = threadIdx.x >> 5, lane = threadIdx.x & 31;
    int p0 = blockIdx.x * 1024;
    int b  = p0 >> 15;
    float tc = fminf(fmaxf(temp[hh], 0.01f), 5.0f);
    float inv_tc = 1.0f / tc;
    float wcol[32];
#pragma unroll
    for (int c = 0; c < 32; ++c) wcol[c] = wsl[c * 32 + lane];  // wslice[c][g=lane]
    float bias_g = bsl[lane];
    float nacc = 0.f;
    float stacc[32];
#pragma unroll
    for (int g = 0; g < 32; ++g) stacc[g] = 0.f;

    for (int p = p0; p < p0 + 1024; ++p) {
        size_t base = (size_t)p * 256 + hh * 32;
        float xv = xm[base + lane];           // lane = c
        float lg = bias_g;                    // lane = g
#pragma unroll
        for (int c = 0; c < 32; ++c) lg += __shfl(xv, c, 32) * wcol[c];
        lg *= inv_tc;
        float mx = lg;
#pragma unroll
        for (int off = 16; off > 0; off >>= 1) mx = fmaxf(mx, __shfl_xor(mx, off, 32));
        float e = __expf(lg - mx);
        float ssum = e;
#pragma unroll
        for (int off = 16; off > 0; off >>= 1) ssum += __shfl_xor(ssum, off, 32);
        float swv = e / ssum;
        sw[base + lane] = swv;
        nacc += swv;
        float fv = fxm[base + lane];          // lane = c
#pragma unroll
        for (int g = 0; g < 32; ++g) stacc[g] += __shfl(swv, g, 32) * fv;
    }
    atomicAdd(&norm[(b * 8 + hh) * 32 + lane], nacc);
    int stbase = (b * 8 + hh) * 1024;         // st[bh][g][c]
#pragma unroll
    for (int g = 0; g < 32; ++g) atomicAdd(&st[stbase + g * 32 + lane], stacc[g]);
}

// ---------------------------------------------------------------------------
// Tiny 32x32 attention over slice tokens, one wave per (b,head).
// ---------------------------------------------------------------------------
__global__ void k_attn(const float* __restrict__ stn, const float* __restrict__ norm,
                       const float* __restrict__ wq, const float* __restrict__ bq,
                       const float* __restrict__ wk, const float* __restrict__ bk,
                       const float* __restrict__ wv, const float* __restrict__ bv,
                       float* __restrict__ token) {
    __shared__ float S[32][33], Q[32][33], Km[32][33], Vm[32][33];
    int bh = blockIdx.x;
    int g  = threadIdx.x;  // 32 threads
    const float* stp = stn + bh * 1024;
    float inv_nrm = 1.0f / (norm[bh * 32 + g] + 1e-5f);
    for (int c = 0; c < 32; ++c) S[g][c] = stp[g * 32 + c] * inv_nrm;
    __syncthreads();
    for (int d = 0; d < 32; ++d) {
        float aq = bq[d], ak = bk[d], av = bv[d];
        for (int c = 0; c < 32; ++c) {
            float s = S[g][c];
            aq += s * wq[c * 32 + d];
            ak += s * wk[c * 32 + d];
            av += s * wv[c * 32 + d];
        }
        Q[g][d] = aq; Km[g][d] = ak; Vm[g][d] = av;
    }
    __syncthreads();
    float e[32], mx = -1e30f;
    const float scale = 0.17677669529663687f;  // 1/sqrt(32)
    for (int j = 0; j < 32; ++j) {
        float a = 0.f;
        for (int d = 0; d < 32; ++d) a += Q[g][d] * Km[j][d];
        e[j] = a * scale;
        mx = fmaxf(mx, e[j]);
    }
    float s = 0.f;
    for (int j = 0; j < 32; ++j) { e[j] = __expf(e[j] - mx); s += e[j]; }
    float inv = 1.f / s;
    for (int d = 0; d < 32; ++d) {
        float a = 0.f;
        for (int j = 0; j < 32; ++j) a += e[j] * Vm[j][d];
        token[bh * 1024 + g * 32 + d] = a * inv;
    }
}

// ---------------------------------------------------------------------------
// De-slice -> f16 (feeds the wo WMMA GEMM):
// out_x[m, h*32+c] = sum_g token[b,h,g,c] * sw[m, h*32+g]
// ---------------------------------------------------------------------------
__global__ void k_deslice(const float* __restrict__ token, const float* __restrict__ sw,
                          _Float16* __restrict__ outx) {
    int m = blockIdx.x;
    int col = threadIdx.x;
    int hh = col >> 5, c = col & 31;
    int b = m >> 15;
    const float* tk = token + (b * 8 + hh) * 1024 + c;  // stride 32 over g
    const float* swr = sw + (size_t)m * 256 + hh * 32;
    float a = 0.f;
#pragma unroll
    for (int g = 0; g < 32; ++g) a += tk[g * 32] * swr[g];
    outx[(size_t)m * 256 + col] = (_Float16)a;
}

// ---------------------------------------------------------------------------
// Final: out[m] = ln3(h[m]) @ w_out + b_out. 8 waves/block, wave per point.
// ---------------------------------------------------------------------------
__global__ void k_final(const float* __restrict__ hbuf, const float* __restrict__ g,
                        const float* __restrict__ bta, const float* __restrict__ wout,
                        const float* __restrict__ bout, float* __restrict__ out) {
    int wave = threadIdx.x >> 5, lane = threadIdx.x & 31;
    int m = blockIdx.x * 8 + wave;
    const float* row = hbuf + (size_t)m * 256;
    float v[8];
    float s = 0.f;
#pragma unroll
    for (int i = 0; i < 8; ++i) { v[i] = row[i * 32 + lane]; s += v[i]; }
#pragma unroll
    for (int off = 16; off > 0; off >>= 1) s += __shfl_xor(s, off, 32);
    float mu = s * (1.f / 256.f);
    float q = 0.f;
#pragma unroll
    for (int i = 0; i < 8; ++i) { float d = v[i] - mu; q += d * d; }
#pragma unroll
    for (int off = 16; off > 0; off >>= 1) q += __shfl_xor(q, off, 32);
    float rs = rsqrtf(q * (1.f / 256.f) + 1e-5f);
    float acc = 0.f;
#pragma unroll
    for (int i = 0; i < 8; ++i) {
        int col = i * 32 + lane;
        acc += ((v[i] - mu) * rs * g[col] + bta[col]) * wout[col];
    }
#pragma unroll
    for (int off = 16; off > 0; off >>= 1) acc += __shfl_xor(acc, off, 32);
    if (lane == 0) out[m] = acc + bout[0];
}

// ---------------------------------------------------------------------------
// Host orchestration
// ---------------------------------------------------------------------------
extern "C" void kernel_launch(void* const* d_in, const int* in_sizes, int n_in,
                              void* d_out, int out_size, void* d_ws, size_t ws_size,
                              hipStream_t stream) {
    (void)in_sizes; (void)n_in; (void)out_size; (void)ws_size;
    const float* X     = (const float*)d_in[0];
    const float* FX    = (const float*)d_in[1];
    const float* preW1 = (const float*)d_in[2];
    const float* preB1 = (const float*)d_in[3];
    const float* preW2 = (const float*)d_in[4];
    const float* preB2 = (const float*)d_in[5];
    const float* ph    = (const float*)d_in[6];

    float* ws   = (float*)d_ws;
    const size_t BUF = (size_t)MROWS * 256;
    float* h    = ws;                  // residual stream (fp32)
    float* bA   = ws + BUF;            // sw
    float* bB   = bA + BUF;            // x_mid
    float* bC   = bB + BUF;            // fx_mid
    float* norm = bC + BUF;            // 1024
    float* st   = norm + 1024;         // 32768
    float* token = st + 32768;         // 32768
    _Float16* a16 = (_Float16*)(token + 32768);  // ln/deslice f16 GEMM input
    _Float16* h16 = a16 + BUF;                   // MLP hidden f16
    _Float16* wt0 = h16 + BUF;                   // transposed f16 weights
    _Float16* wt1 = wt0 + 65536;
    _Float16* wt2 = wt1 + 65536;
    _Float16* wt3 = wt2 + 65536;
    _Float16* wt4 = wt3 + 65536;

    dim3 gemmGrid(MROWS / 128, HDIM / 64);

    k_pre<<<MROWS, 256, 0, stream>>>(X, FX, preW1, preB1, preW2, preB2, ph, h);

    for (int l = 0; l < 5; ++l) {
        int base = 7 + 23 * l;
        const float* ln1_g  = (const float*)d_in[base + 0];
        const float* ln1_b  = (const float*)d_in[base + 1];
        const float* wx     = (const float*)d_in[base + 2];
        const float* bx     = (const float*)d_in[base + 3];
        const float* wfx    = (const float*)d_in[base + 4];
        const float* bfx    = (const float*)d_in[base + 5];
        const float* wsl    = (const float*)d_in[base + 6];
        const float* bsl    = (const float*)d_in[base + 7];
        const float* temp   = (const float*)d_in[base + 8];
        const float* wq     = (const float*)d_in[base + 9];
        const float* bq     = (const float*)d_in[base + 10];
        const float* wk     = (const float*)d_in[base + 11];
        const float* bk     = (const float*)d_in[base + 12];
        const float* wv     = (const float*)d_in[base + 13];
        const float* bv     = (const float*)d_in[base + 14];
        const float* wo     = (const float*)d_in[base + 15];
        const float* bo     = (const float*)d_in[base + 16];
        const float* ln2_g  = (const float*)d_in[base + 17];
        const float* ln2_b  = (const float*)d_in[base + 18];
        const float* mw1    = (const float*)d_in[base + 19];
        const float* mb1    = (const float*)d_in[base + 20];
        const float* mw2    = (const float*)d_in[base + 21];
        const float* mb2    = (const float*)d_in[base + 22];

        // transposed f16 weights for this layer
        k_wt<<<256, 256, 0, stream>>>(wx,  wt0);
        k_wt<<<256, 256, 0, stream>>>(wfx, wt1);
        k_wt<<<256, 256, 0, stream>>>(wo,  wt2);
        k_wt<<<256, 256, 0, stream>>>(mw1, wt3);
        k_wt<<<256, 256, 0, stream>>>(mw2, wt4);

        k_ln<<<MROWS / 8, 256, 0, stream>>>(h, ln1_g, ln1_b, a16);
        k_gemm<<<gemmGrid, 256, 0, stream>>>(a16, wt0, bx,  nullptr, bB, nullptr, 0);
        k_gemm<<<gemmGrid, 256, 0, stream>>>(a16, wt1, bfx, nullptr, bC, nullptr, 0);
        hipMemsetAsync(norm, 0, (1024 + 32768) * sizeof(float), stream);
        k_slice<<<MROWS / 1024, 256, 0, stream>>>(bB, bC, wsl, bsl, temp, bA, norm, st);
        k_attn<<<32, 32, 0, stream>>>(st, norm, wq, bq, wk, bk, wv, bv, token);
        k_deslice<<<MROWS, 256, 0, stream>>>(token, bA, a16);
        k_gemm<<<gemmGrid, 256, 0, stream>>>(a16, wt2, bo, h, h, nullptr, 0);
        k_ln<<<MROWS / 8, 256, 0, stream>>>(h, ln2_g, ln2_b, a16);
        k_gemm<<<gemmGrid, 256, 0, stream>>>(a16, wt3, mb1, nullptr, nullptr, h16, 1);
        k_gemm<<<gemmGrid, 256, 0, stream>>>(h16, wt4, mb2, h, h, nullptr, 0);
    }

    int lb = 7 + 23 * 4;
    const float* ln3_g = (const float*)d_in[lb + 23];
    const float* ln3_b = (const float*)d_in[lb + 24];
    const float* wout  = (const float*)d_in[lb + 25];
    const float* bout  = (const float*)d_in[lb + 26];
    k_final<<<MROWS / 8, 256, 0, stream>>>(h, ln3_g, ln3_b, wout, bout, (float*)d_out);
}
